// FNOGNO_3917010174476
// MI455X (gfx1250) — matrix operand order repaired and checked by reading, not compiled, not run on hardware
//
#include <hip/hip_runtime.h>
#include <hip/hip_bf16.h>
#include <stdint.h>

// ---------------------------------------------------------------------------
// FNO+GNO for MI455X (gfx1250, wave32, WMMA).
// Dense layers via v_wmma_f32_16x16x32_f16 with 2x2 wave tiling (4 WMMAs per
// K-step, 2x register reuse of A/B frags) and LDS-staged B panels. B staging
// uses CDNA5 async global->LDS copies (ASYNCcnt) when the toolchain exposes
// them, else plain global_load+ds_store. 3D FFT as separable 32-point DFT
// passes in fp32 VALU (L2-resident). Neighbor search: wave32 ballot/prefix.
// ---------------------------------------------------------------------------

typedef __attribute__((ext_vector_type(16))) _Float16 v16h;
typedef __attribute__((ext_vector_type(8)))  _Float16 v8h;
typedef __attribute__((ext_vector_type(8)))  float    v8f;
typedef int i32x4 __attribute__((vector_size(4 * sizeof(int))));

#if defined(__has_builtin)
#if __has_builtin(__builtin_amdgcn_global_load_async_to_lds_b128)
#define HAVE_ASYNC_LDS 1
#endif
#endif

#define ASG __attribute__((address_space(1)))
#define ASL __attribute__((address_space(3)))

#define NIN   32768      // 32^3 grid points
#define NOUTQ 4096       // query points
#define HCH   64         // hidden channels
#define KNB   48         // neighbor cap
#define QCHUNK 512       // queries per GNO chunk
#define NCHUNK (NOUTQ / QCHUNK)
#define CHROWS (QCHUNK * KNB)   // 24576 rows per GNO chunk
#define R2    (0.06f * 0.06f)

__device__ __forceinline__ float gelu_f(float x) {
    // tanh approximation (jax.nn.gelu default)
    float x3 = x * x * x;
    return 0.5f * x * (1.0f + tanhf(0.7978845608028654f * (x + 0.044715f * x3)));
}

// A 16x32 f16 fragment: lane = (g, r); row = r, K split per ISA 7.12.2
__device__ __forceinline__ v16h frag_a(const _Float16* __restrict__ p, int kb, int g) {
    v8h lo = *(const v8h*)(p + kb + g * 8);
    v8h hi = *(const v8h*)(p + kb + g * 8 + 16);
    return __builtin_shufflevector(lo, hi, 0,1,2,3,4,5,6,7,8,9,10,11,12,13,14,15);
}
// B 32x16 f16 fragment from K-contiguous (transposed) storage
__device__ __forceinline__ v16h frag_b(const _Float16* p, int kb, int g) {
    v8h lo = *(const v8h*)(p + kb + g * 8);
    v8h hi = *(const v8h*)(p + kb + g * 8 + 16);
    return __builtin_shufflevector(lo, hi, 0,1,2,3,4,5,6,7,8,9,10,11,12,13,14,15);
}
__device__ __forceinline__ v8f wmma16(v16h a, v16h b, v8f c) {
    return __builtin_amdgcn_wmma_f32_16x16x32_f16(false, a, false, b, (short)0, c, false, false);
}

// ---------------------------------------------------------------------------
// WMMA GEMM:  C[M,N] = A[M,K] * B[K,N] (+bias) (+gelu)
//   A  : f16 row-major, K multiple of 32; M multiple of 128
//   BT : f16, B transposed -> [N][K] (K contiguous)
//   Oh : f16 output [M][N] (feeds next GEMM's A)
//   Of : f32 output [M][ldcf], only cols < ncols_f32 stored
// Block = 128 threads (4 waves). Block covers 128(M) x 16*NSUB (N); each wave
// owns a 32 x 16*NSUB tile = 2xNSUB WMMA accumulators. B panel staged once
// into LDS (async global->LDS where available).
// NSUB is a compile-time constant so the K-loop is branch-free.
// ---------------------------------------------------------------------------
template<int NSUB, bool GELU_EP, bool WRITE_F16, bool WRITE_F32>
__global__ __launch_bounds__(128)
void gemm_f16_wmma(const _Float16* __restrict__ A,
                   const _Float16* __restrict__ BT,
                   const float* __restrict__ bias, int nbias,
                   _Float16* __restrict__ Oh,
                   float* __restrict__ Of,
                   int M, int N, int Kd, int ldcf, int ncols_f32)
{
    extern __shared__ _Float16 sb[];               // [32][Kd] B panel
    const int lane = threadIdx.x & 31;
    const int wave = threadIdx.x >> 5;
    const int tn   = blockIdx.x;                   // 32-wide N panel
    const int wm   = blockIdx.y * 4 + wave;        // 32-wide M panel per wave
    const int ncol0 = tn * 32;

    // ---- cooperative stage of B panel (32 cols x Kd halves) into LDS ----
    {
        const int kch = Kd >> 3;                   // 16B chunks per column
        const int nch = 32 * kch;
        for (int t = threadIdx.x; t < nch; t += 128) {
            int col = t / kch, k8 = (t % kch) * 8;
            _Float16* dst = sb + (size_t)col * Kd + k8;
            if (ncol0 + col < N) {
                const _Float16* src = BT + (size_t)(ncol0 + col) * Kd + k8;
#ifdef HAVE_ASYNC_LDS
                // CDNA5 async copy: GLOBAL_LOAD_ASYNC_TO_LDS_B128 (ASYNCcnt)
                __builtin_amdgcn_global_load_async_to_lds_b128(
                    (ASG i32x4*)(uintptr_t)src,
                    (ASL i32x4*)(unsigned)(uintptr_t)dst, 0, 0);
#else
                *(v8h*)dst = *(const v8h*)src;
#endif
            } else {
                v8h z = {};
                *(v8h*)dst = z;
            }
        }
#ifdef HAVE_ASYNC_LDS
#if __has_builtin(__builtin_amdgcn_s_wait_asynccnt)
        __builtin_amdgcn_s_wait_asynccnt(0);
#else
        asm volatile("s_wait_asynccnt 0x0" ::: "memory");
#endif
#endif
    }
    __syncthreads();
    if (wm * 32 >= M) return;

    const int g = lane >> 4;
    const int r = lane & 15;
    const _Float16* __restrict__ Ap0 = A + (size_t)(wm * 32 + r) * Kd;
    const _Float16* __restrict__ Ap1 = A + (size_t)(wm * 32 + 16 + r) * Kd;
    const _Float16* Bl0 = sb + (size_t)r * Kd;
    const _Float16* Bl1 = sb + (size_t)(16 + r) * Kd;

    v8f acc00 = {}, acc01 = {}, acc10 = {}, acc11 = {};
    for (int kb = 0; kb < Kd; kb += 32) {
        __builtin_prefetch(Ap0 + kb + 64, 0, 1);      // global_prefetch_b8
        __builtin_prefetch(Ap1 + kb + 64, 0, 1);
        v16h a0 = frag_a(Ap0, kb, g);
        v16h a1 = frag_a(Ap1, kb, g);
        v16h b0 = frag_b(Bl0, kb, g);                 // ds_load_b128
        acc00 = wmma16(a0, b0, acc00);
        acc10 = wmma16(a1, b0, acc10);
        if (NSUB == 2) {                              // compile-time
            v16h b1 = frag_b(Bl1, kb, g);
            acc01 = wmma16(a0, b1, acc01);
            acc11 = wmma16(a1, b1, acc11);
        }
    }

    auto store_tile = [&](const v8f& acc, int msub, int nsub) {
        int colg = ncol0 + nsub * 16 + r;             // C/D: col = lane&15
        #pragma unroll
        for (int v = 0; v < 8; ++v) {
            int row = wm * 32 + msub * 16 + v + 8 * g; // C/D: row = v + 8*group
            float val = acc[v];
            if (bias && colg < nbias) val += bias[colg];
            if (GELU_EP) val = gelu_f(val);
            if (WRITE_F16) Oh[(size_t)row * N + colg] = (_Float16)val;
            if (WRITE_F32 && colg < ncols_f32) Of[(size_t)row * ldcf + colg] = val;
        }
    };
    store_tile(acc00, 0, 0);
    store_tile(acc10, 1, 0);
    if (NSUB == 2) {
        store_tile(acc01, 0, 1);
        store_tile(acc11, 1, 1);
    }
}

// ---------------------------------------------------------------------------
// Weight prep: w[K][N] fp32 -> wt[Npad][Kpad] f16 (transposed, zero padded)
// ---------------------------------------------------------------------------
__global__ void prep_w(const float* __restrict__ w, _Float16* __restrict__ wt,
                       int K, int N, int Kp, int Np)
{
    int t = blockIdx.x * blockDim.x + threadIdx.x;
    if (t >= Kp * Np) return;
    int n = t / Kp, k = t % Kp;
    float v = (k < K && n < N) ? w[(size_t)k * N + n] : 0.0f;
    wt[(size_t)n * Kp + k] = (_Float16)v;
}

// Build lift input A0[NIN][32] f16 = [f(3), in_p(3), 0-pad]
__global__ void build_x0(const float* __restrict__ f, const float* __restrict__ in_p,
                         _Float16* __restrict__ a0)
{
    int s = blockIdx.x * blockDim.x + threadIdx.x;
    if (s >= NIN) return;
    _Float16* o = a0 + (size_t)s * 32;
    o[0] = (_Float16)f[s*3+0]; o[1] = (_Float16)f[s*3+1]; o[2] = (_Float16)f[s*3+2];
    o[3] = (_Float16)in_p[s*3+0]; o[4] = (_Float16)in_p[s*3+1]; o[5] = (_Float16)in_p[s*3+2];
    #pragma unroll
    for (int t = 6; t < 32; ++t) o[t] = (_Float16)0.0f;
}

// ---------------------------------------------------------------------------
// Separable 32-point DFT pass along one axis of [32,32,32,64] complex field.
// sign=-1, scale=1/32 forward; sign=+1, scale=1 inverse. iim==null -> imag 0.
// ---------------------------------------------------------------------------
__global__ void dft32(const float* __restrict__ ire, const float* __restrict__ iim,
                      float* __restrict__ ore, float* __restrict__ oim,
                      int axis, float sign, float scale)
{
    int e = blockIdx.x * blockDim.x + threadIdx.x;
    if (e >= NIN * HCH) return;
    int c = e & 63;
    int s = e >> 6;
    int z = s & 31, y = (s >> 5) & 31, x = s >> 10;
    int k, stride, base;
    if (axis == 0)      { k = x; stride = 32 * 32 * 64; base = (y * 32 + z) * 64 + c; }
    else if (axis == 1) { k = y; stride = 32 * 64;      base = (x * 1024 + z) * 64 + c; }
    else                { k = z; stride = 64;           base = (x * 1024 + y * 32) * 64 + c; }
    float ar = 0.0f, ai = 0.0f;
    float w = sign * 6.2831853071795864f * (float)k * (1.0f / 32.0f);
    for (int n = 0; n < 32; ++n) {
        float sn, cs;
        __sincosf(w * (float)n, &sn, &cs);
        int id = base + n * stride;
        float vr = ire[id];
        float vi = iim ? iim[id] : 0.0f;
        ar += vr * cs - vi * sn;
        ai += vr * sn + vi * cs;
    }
    ore[e] = ar * scale;
    oim[e] = ai * scale;
}

// ---------------------------------------------------------------------------
// Spectral corner multiply: out[k,:] = spec_w[l,ci,:,:,m]^T * xf[k,:]
// (complex 64x64 matvec per mode point), plus Hermitian mirror for kz>0
// so the full-complex inverse DFT reproduces irfftn.
// ---------------------------------------------------------------------------
__global__ __launch_bounds__(64)
void spectral(const float* __restrict__ xr, const float* __restrict__ xi,
              const float* __restrict__ specw,   // complex64 interleaved
              float* __restrict__ outr, float* __restrict__ outi, int layer)
{
    __shared__ float sr[64], si[64];
    int b  = blockIdx.x;              // 4 corners * 512 modes
    int ci = b >> 9;
    int m  = b & 511;
    int mx = m >> 6, my = (m >> 3) & 7, mz = m & 7;
    int kx = (ci & 2) ? (24 + mx) : mx;
    int ky = (ci & 1) ? (24 + my) : my;
    int kz = mz;
    int o  = threadIdx.x;
    int inb = ((kx * 32 + ky) * 32 + kz) * 64;
    sr[o] = xr[inb + o];
    si[o] = xi[inb + o];
    __syncthreads();
    float ar = 0.0f, ai = 0.0f;
    size_t wb = (size_t)(layer * 4 + ci) * 64;
    for (int i = 0; i < 64; ++i) {
        size_t wi = ((((wb + i) * 64 + o) * 8 + mx) * 8 + my) * 8 + mz;
        float wr = specw[2 * wi], wim = specw[2 * wi + 1];
        float vr = sr[i], vi = si[i];
        ar += vr * wr - vi * wim;
        ai += vr * wim + vi * wr;
    }
    outr[inb + o] = ar;
    outi[inb + o] = ai;
    if (kz != 0) {   // Hermitian mirror (kz' = 32-kz never written directly)
        int mkx = (32 - kx) & 31, mky = (32 - ky) & 31, mkz = 32 - kz;
        int ob2 = ((mkx * 32 + mky) * 32 + mkz) * 64 + o;
        outr[ob2] = ar;
        outi[ob2] = -ai;
    }
}

// x_new = Re(idft) + spec_b[l] + (skip conv incl. skip_b) ; gelu if not last
__global__ void fno_combine(const float* __restrict__ fre,
                            const float* __restrict__ skipo,
                            const float* __restrict__ specb,
                            float* __restrict__ lat32, _Float16* __restrict__ lat16,
                            int layer, int dogelu)
{
    int e = blockIdx.x * blockDim.x + threadIdx.x;
    if (e >= NIN * HCH) return;
    int c = e & 63;
    float v = fre[e] + specb[layer * 64 + c] + skipo[e];
    if (dogelu) v = gelu_f(v);
    lat32[e] = v;
    lat16[e] = (_Float16)v;
}

// ---------------------------------------------------------------------------
// Neighbor search: one wave32 per query. Deterministically collects the
// first KNB in-radius grid points in index order via ballot + prefix popc.
// ---------------------------------------------------------------------------
__global__ __launch_bounds__(32)
void neighbors_k(const float* __restrict__ yp, const float* __restrict__ xq,
                 int* __restrict__ idx, int* __restrict__ cnt)
{
    int q = blockIdx.x;
    int lane = threadIdx.x;
    float qx = xq[q*3+0], qy = xq[q*3+1], qz = xq[q*3+2];
    int total = 0;
    for (int base = 0; base < NIN && total < KNB; base += 32) {
        int i = base + lane;
        float dx = yp[i*3+0] - qx;
        float dy = yp[i*3+1] - qy;
        float dz = yp[i*3+2] - qz;
        bool hit = (dx*dx + dy*dy + dz*dz) < R2;
        unsigned mask = (unsigned)__ballot(hit);
        int prefix = __popc(mask & ((1u << lane) - 1u));
        int slot = total + prefix;
        if (hit && slot < KNB) idx[q * KNB + slot] = i;
        total += __popc(mask);
    }
    total = total > KNB ? KNB : total;
    for (int s2 = total + lane; s2 < KNB; s2 += 32) idx[q * KNB + s2] = 0;
    if (lane == 0) cnt[q] = total;
}

// GNO MLP input features [rows][32] f16 = [y[idx](3), xq(3), 0-pad]
__global__ void feats_k(const float* __restrict__ yp, const float* __restrict__ xq,
                        const int* __restrict__ idx, _Float16* __restrict__ feats,
                        int q0, int nrows)
{
    int rl = blockIdx.x * blockDim.x + threadIdx.x;
    if (rl >= nrows) return;
    int q = q0 + rl / KNB;
    int s = rl % KNB;
    int i = idx[q * KNB + s];
    _Float16* o = feats + (size_t)rl * 32;
    o[0] = (_Float16)yp[i*3+0]; o[1] = (_Float16)yp[i*3+1]; o[2] = (_Float16)yp[i*3+2];
    o[3] = (_Float16)xq[q*3+0]; o[4] = (_Float16)xq[q*3+1]; o[5] = (_Float16)xq[q*3+2];
    #pragma unroll
    for (int t = 6; t < 32; ++t) o[t] = (_Float16)0.0f;
}

// gno[q,c] = (1/cnt) * sum_s k3[q,s,c] * latent[idx[q,s], c]
__global__ void gno_reduce(const float* __restrict__ k3, const int* __restrict__ idx,
                           const int* __restrict__ cnt, const float* __restrict__ lat,
                           float* __restrict__ go32, _Float16* __restrict__ go16,
                           int q0, int nq)
{
    int t = blockIdx.x * blockDim.x + threadIdx.x;
    if (t >= nq * HCH) return;
    int ql = t >> 6, c = t & 63;
    int q = q0 + ql;
    int n = cnt[q];
    float acc = 0.0f;
    for (int s = 0; s < n; ++s) {
        int i = idx[q * KNB + s];
        acc += k3[((size_t)(ql * KNB + s)) * HCH + c] * lat[(size_t)i * HCH + c];
    }
    float v = acc / (n > 0 ? (float)n : 1.0f);
    go32[(size_t)q * HCH + c] = v;
    go16[(size_t)q * HCH + c] = (_Float16)v;
}

// ---------------------------------------------------------------------------
// Workspace arena (~124 MB)
// ---------------------------------------------------------------------------
static constexpr size_t AL(size_t x) { return (x + 255) & ~size_t(255); }
static constexpr size_t O_WT_L1 = 0;
static constexpr size_t O_WT_L2 = O_WT_L1 + AL(256*32*2);
static constexpr size_t O_WT_SK = O_WT_L2 + AL(64*256*2);
static constexpr size_t O_WT_G1 = O_WT_SK + AL(4*64*64*2);
static constexpr size_t O_WT_G2 = O_WT_G1 + AL(512*32*2);
static constexpr size_t O_WT_G3 = O_WT_G2 + AL(256*512*2);
static constexpr size_t O_WT_P1 = O_WT_G3 + AL(64*256*2);
static constexpr size_t O_WT_P2 = O_WT_P1 + AL(256*64*2);
static constexpr size_t O_A0    = O_WT_P2 + AL(16*256*2);
static constexpr size_t O_ACT1  = O_A0    + AL((size_t)NIN*32*2);
static constexpr size_t O_LAT32 = O_ACT1  + AL((size_t)NIN*256*2);
static constexpr size_t O_LAT16 = O_LAT32 + AL((size_t)NIN*HCH*4);
static constexpr size_t O_CRA   = O_LAT16 + AL((size_t)NIN*HCH*2);
static constexpr size_t O_CIA   = O_CRA   + AL((size_t)NIN*HCH*4);
static constexpr size_t O_CRB   = O_CIA   + AL((size_t)NIN*HCH*4);
static constexpr size_t O_CIB   = O_CRB   + AL((size_t)NIN*HCH*4);
static constexpr size_t O_SKIP  = O_CIB   + AL((size_t)NIN*HCH*4);
static constexpr size_t O_IDX   = O_SKIP  + AL((size_t)NIN*HCH*4);
static constexpr size_t O_CNT   = O_IDX   + AL((size_t)NOUTQ*KNB*4);
static constexpr size_t O_FEATS = O_CNT   + AL((size_t)NOUTQ*4);
static constexpr size_t O_G1    = O_FEATS + AL((size_t)CHROWS*32*2);
static constexpr size_t O_G2    = O_G1    + AL((size_t)CHROWS*512*2);
static constexpr size_t O_K3    = O_G2    + AL((size_t)CHROWS*256*2);
static constexpr size_t O_GNO32 = O_K3    + AL((size_t)CHROWS*HCH*4);
static constexpr size_t O_GNO16 = O_GNO32 + AL((size_t)NOUTQ*HCH*4);
static constexpr size_t O_PACT  = O_GNO16 + AL((size_t)NOUTQ*HCH*2);

extern "C" void kernel_launch(void* const* d_in, const int* in_sizes, int n_in,
                              void* d_out, int out_size, void* d_ws, size_t ws_size,
                              hipStream_t stream) {
    (void)in_sizes; (void)n_in; (void)out_size; (void)ws_size;
    const float* in_p    = (const float*)d_in[0];
    const float* out_p   = (const float*)d_in[1];
    const float* f       = (const float*)d_in[2];
    const float* lift_w1 = (const float*)d_in[3];
    const float* lift_b1 = (const float*)d_in[4];
    const float* lift_w2 = (const float*)d_in[5];
    const float* lift_b2 = (const float*)d_in[6];
    const float* spec_w  = (const float*)d_in[7];   // complex64 interleaved
    const float* spec_b  = (const float*)d_in[8];
    const float* skip_w  = (const float*)d_in[9];
    const float* skip_b  = (const float*)d_in[10];
    const float* gno_w1  = (const float*)d_in[11];
    const float* gno_b1  = (const float*)d_in[12];
    const float* gno_w2  = (const float*)d_in[13];
    const float* gno_b2  = (const float*)d_in[14];
    const float* gno_w3  = (const float*)d_in[15];
    const float* gno_b3  = (const float*)d_in[16];
    const float* proj_w1 = (const float*)d_in[17];
    const float* proj_b1 = (const float*)d_in[18];
    const float* proj_w2 = (const float*)d_in[19];
    const float* proj_b2 = (const float*)d_in[20];

    char* ws = (char*)d_ws;
    _Float16* WT_L1 = (_Float16*)(ws + O_WT_L1);
    _Float16* WT_L2 = (_Float16*)(ws + O_WT_L2);
    _Float16* WT_SK = (_Float16*)(ws + O_WT_SK);
    _Float16* WT_G1 = (_Float16*)(ws + O_WT_G1);
    _Float16* WT_G2 = (_Float16*)(ws + O_WT_G2);
    _Float16* WT_G3 = (_Float16*)(ws + O_WT_G3);
    _Float16* WT_P1 = (_Float16*)(ws + O_WT_P1);
    _Float16* WT_P2 = (_Float16*)(ws + O_WT_P2);
    _Float16* A0    = (_Float16*)(ws + O_A0);
    _Float16* ACT1  = (_Float16*)(ws + O_ACT1);
    float*    LAT32 = (float*)   (ws + O_LAT32);
    _Float16* LAT16 = (_Float16*)(ws + O_LAT16);
    float*    CRA   = (float*)   (ws + O_CRA);
    float*    CIA   = (float*)   (ws + O_CIA);
    float*    CRB   = (float*)   (ws + O_CRB);
    float*    CIB   = (float*)   (ws + O_CIB);
    float*    SKIPO = (float*)   (ws + O_SKIP);
    int*      IDX   = (int*)     (ws + O_IDX);
    int*      CNT   = (int*)     (ws + O_CNT);
    _Float16* FEATS = (_Float16*)(ws + O_FEATS);
    _Float16* G1    = (_Float16*)(ws + O_G1);
    _Float16* G2    = (_Float16*)(ws + O_G2);
    float*    K3    = (float*)   (ws + O_K3);
    float*    GNO32 = (float*)   (ws + O_GNO32);
    _Float16* GNO16 = (_Float16*)(ws + O_GNO16);
    _Float16* PACT  = (_Float16*)(ws + O_PACT);

    auto prep = [&](const float* w, _Float16* wt, int K, int N, int Kp, int Np) {
        int tot = Kp * Np;
        prep_w<<<(tot + 255) / 256, 256, 0, stream>>>(w, wt, K, N, Kp, Np);
    };
    // weight conversion (fp32 -> f16, transposed, padded)
    prep(lift_w1, WT_L1, 6, 256, 32, 256);
    prep(lift_w2, WT_L2, 256, 64, 256, 64);
    for (int l = 0; l < 4; ++l)
        prep(skip_w + (size_t)l * 64 * 64, WT_SK + (size_t)l * 64 * 64, 64, 64, 64, 64);
    prep(gno_w1, WT_G1, 6, 512, 32, 512);
    prep(gno_w2, WT_G2, 512, 256, 512, 256);
    prep(gno_w3, WT_G3, 256, 64, 256, 64);
    prep(proj_w1, WT_P1, 64, 256, 64, 256);
    prep(proj_w2, WT_P2, 256, 1, 256, 16);

    const dim3 blk128(128);
    // block covers 128 (M) x 32 (N); LDS = B panel (32 cols x Kd halves)
    auto ggrid = [](int M, int N) { return dim3((N + 31) / 32, M / 128); };
    auto gsh   = [](int Kd) { return (size_t)Kd * 32 * sizeof(_Float16); };

    // ---- Lift: [NIN,6]->256 (gelu)->64 ----
    build_x0<<<(NIN + 255) / 256, 256, 0, stream>>>(f, in_p, A0);
    gemm_f16_wmma<2, true, true, false><<<ggrid(NIN, 256), blk128, gsh(32), stream>>>(
        A0, WT_L1, lift_b1, 256, ACT1, nullptr, NIN, 256, 32, 0, 0);
    gemm_f16_wmma<2, false, true, true><<<ggrid(NIN, 64), blk128, gsh(256), stream>>>(
        ACT1, WT_L2, lift_b2, 64, LAT16, LAT32, NIN, 64, 256, 64, 64);

    // ---- FNO layers ----
    const int NTOT = NIN * HCH;
    const int DBLK = (NTOT + 255) / 256;
    for (int l = 0; l < 4; ++l) {
        // skip conv (WMMA) with bias
        gemm_f16_wmma<2, false, false, true><<<ggrid(NIN, 64), blk128, gsh(64), stream>>>(
            LAT16, WT_SK + (size_t)l * 64 * 64, skip_b + l * 64, 64,
            nullptr, SKIPO, NIN, 64, 64, 64, 64);
        // forward DFT (norm='forward': 1/32 per axis)
        dft32<<<DBLK, 256, 0, stream>>>(LAT32, nullptr, CRA, CIA, 0, -1.0f, 1.0f / 32.0f);
        dft32<<<DBLK, 256, 0, stream>>>(CRA, CIA, CRB, CIB, 1, -1.0f, 1.0f / 32.0f);
        dft32<<<DBLK, 256, 0, stream>>>(CRB, CIB, CRA, CIA, 2, -1.0f, 1.0f / 32.0f);
        // spectral corners into zeroed full-complex spectrum (+ Hermitian mirror)
        (void)hipMemsetAsync(CRB, 0, (size_t)NTOT * 4, stream);
        (void)hipMemsetAsync(CIB, 0, (size_t)NTOT * 4, stream);
        spectral<<<2048, 64, 0, stream>>>(CRA, CIA, spec_w, CRB, CIB, l);
        // inverse DFT (no scaling)
        dft32<<<DBLK, 256, 0, stream>>>(CRB, CIB, CRA, CIA, 0, 1.0f, 1.0f);
        dft32<<<DBLK, 256, 0, stream>>>(CRA, CIA, CRB, CIB, 1, 1.0f, 1.0f);
        dft32<<<DBLK, 256, 0, stream>>>(CRB, CIB, CRA, CIA, 2, 1.0f, 1.0f);
        // combine + activation
        fno_combine<<<DBLK, 256, 0, stream>>>(CRA, SKIPO, spec_b, LAT32, LAT16,
                                              l, (l < 3) ? 1 : 0);
    }

    // ---- GNO ----
    neighbors_k<<<NOUTQ, 32, 0, stream>>>(in_p, out_p, IDX, CNT);
    for (int ch = 0; ch < NCHUNK; ++ch) {
        int q0 = ch * QCHUNK;
        feats_k<<<(CHROWS + 255) / 256, 256, 0, stream>>>(in_p, out_p, IDX, FEATS, q0, CHROWS);
        gemm_f16_wmma<2, true, true, false><<<ggrid(CHROWS, 512), blk128, gsh(32), stream>>>(
            FEATS, WT_G1, gno_b1, 512, G1, nullptr, CHROWS, 512, 32, 0, 0);
        gemm_f16_wmma<2, true, true, false><<<ggrid(CHROWS, 256), blk128, gsh(512), stream>>>(
            G1, WT_G2, gno_b2, 256, G2, nullptr, CHROWS, 256, 512, 0, 0);
        gemm_f16_wmma<2, false, false, true><<<ggrid(CHROWS, 64), blk128, gsh(256), stream>>>(
            G2, WT_G3, gno_b3, 64, nullptr, K3, CHROWS, 64, 256, 64, 64);
        gno_reduce<<<(QCHUNK * HCH + 255) / 256, 256, 0, stream>>>(
            K3, IDX, CNT, LAT32, GNO32, GNO16, q0, QCHUNK);
    }

    // ---- Projection: 64 -> 256 (gelu) -> 1 ----
    gemm_f16_wmma<2, true, true, false><<<ggrid(NOUTQ, 256), blk128, gsh(64), stream>>>(
        GNO16, WT_P1, proj_b1, 256, PACT, nullptr, NOUTQ, 256, 64, 0, 0);
    gemm_f16_wmma<1, false, false, true><<<ggrid(NOUTQ, 16), blk128, gsh(256), stream>>>(
        PACT, WT_P2, proj_b2, 1, nullptr, (float*)d_out, NOUTQ, 16, 256, 1, 1);
}
